// LatticeSelfAttnLayer_30562987278772
// MI455X (gfx1250) — compile-verified, matrix operand order checked by reading, not run on hardware
//
#include <hip/hip_runtime.h>
#include <hip/hip_bf16.h>
#include <math.h>

typedef __attribute__((ext_vector_type(16))) __bf16 v16bf;
typedef __attribute__((ext_vector_type(8)))  __bf16 v8bf;
typedef __attribute__((ext_vector_type(8)))  float  v8f;

// ---------------------------------------------------------------------------
// WMMA helper: D = A(16x32 bf16) * B(32x16 bf16) + C(16x16 f32)
// ---------------------------------------------------------------------------
__device__ inline v8f wmma_bf16(v16bf a, v16bf b, v8f c) {
  return __builtin_amdgcn_wmma_f32_16x16x32_bf16(
      /*neg_a=*/false, a, /*neg_b=*/false, b,
      /*c_mod=*/(short)0, c, /*reuse_a=*/false, /*reuse_b=*/false);
}

// Load one lane's A/B fragment: two contiguous 8-element (16B) chunks,
// second chunk +16 elements along K. `base` already includes row + g*8 offset.
__device__ inline v16bf ld_frag(const __bf16* base) {
  v8bf lo = *(const v8bf*)(base);
  v8bf hi = *(const v8bf*)(base + 16);
  return __builtin_shufflevector(lo, hi, 0,1,2,3,4,5,6,7,8,9,10,11,12,13,14,15);
}

// ---------------------------------------------------------------------------
// CDNA5 async global->LDS copy (16 bytes / lane), tracked by ASYNCcnt.
// GVS mode: mem = SADDR(u64) + VADDR(u32) ; VDST = per-lane LDS byte address.
// ---------------------------------------------------------------------------
__device__ inline void async_copy16(unsigned lds_addr, unsigned gbyte_off,
                                    const void* sbase) {
  asm volatile("global_load_async_to_lds_b128 %0, %1, %2"
               :: "v"(lds_addr), "v"(gbyte_off), "s"(sbase)
               : "memory");
}
__device__ inline void wait_async() {
  asm volatile("s_wait_asynccnt 0x0" ::: "memory");
}

// ---------------------------------------------------------------------------
// Conversion kernels
// ---------------------------------------------------------------------------
__global__ __launch_bounds__(256) void cvt_f32_bf16(const float* __restrict__ in,
                                                    __bf16* __restrict__ out, size_t n) {
  size_t i = (size_t)blockIdx.x * 256 + threadIdx.x;
  if (i < n) out[i] = (__bf16)in[i];
}

// W[K][N] (f32, row-major) -> WT[N][K] (bf16) so B-fragments are contiguous in K.
__global__ __launch_bounds__(256) void transpose_w_bf16(const float* __restrict__ W,
                                                        __bf16* __restrict__ WT,
                                                        int K, int N) {
  size_t t = (size_t)blockIdx.x * 256 + threadIdx.x;
  if (t >= (size_t)K * N) return;
  int n = (int)(t % N);
  int k = (int)(t / N);
  WT[(size_t)n * K + k] = (__bf16)W[t];
}

// ---------------------------------------------------------------------------
// bf16 WMMA GEMM. Block tile 128(M) x 128(N), K-step 32, 8 waves in a 4x2
// grid: wave = 32 rows x 64 cols = 2 A-frags x 4 B-frags -> 8 WMMAs/k-step.
// A: [M][K] bf16 row-major.  WT: [N][K] bf16 row-major (B transposed).
// Staging uses global_load_async_to_lds_b128 (ASYNCcnt).
// EPI: 0 = f32 out (+bias), 2 = exact GELU -> bf16 out, 3 = QKV split/scale.
// ---------------------------------------------------------------------------
#define EPI_F32  0
#define EPI_GELU 2
#define EPI_QKV  3

template <int EPI>
__global__ __launch_bounds__(256) void gemm_bf16_wmma(
    const __bf16* __restrict__ A, const __bf16* __restrict__ WT,
    const float* __restrict__ bias, int M, int N, int K,
    float* __restrict__ outF, __bf16* __restrict__ outB,
    __bf16* __restrict__ qOut, __bf16* __restrict__ kOut, __bf16* __restrict__ vOut) {
  __shared__ __bf16 As[128 * 32];
  __shared__ __bf16 Bs[128 * 32];

  const int tid  = threadIdx.x;
  const int wave = tid >> 5;
  const int lane = tid & 31;
  const int g    = lane >> 4;   // K-chunk group
  const int ln   = lane & 15;
  const int wr   = (wave & 3) * 32;   // wave row base within tile
  const int wc   = (wave >> 2) * 64;  // wave col base within tile
  const int m0   = blockIdx.x * 128;
  const int n0   = blockIdx.y * 128;

  const unsigned asBase = (unsigned)(uintptr_t)&As[0];
  const unsigned bsBase = (unsigned)(uintptr_t)&Bs[0];

  // Per-thread staging chunk coordinates (128x32 tile = 512 x 16B chunks)
  const int srow = tid >> 2;            // 0..63
  const int soff = (tid & 3) * 8;       // 0,8,16,24 elements

  v8f acc[2][4] = {};

  for (int k0 = 0; k0 < K; k0 += 32) {
    __syncthreads();
    // Async-stage A tile (rows srow, srow+64) and B tile (rows srow, srow+64)
    {
      unsigned lofs = (unsigned)((srow * 32 + soff) * 2);
      unsigned lofs2 = (unsigned)(((srow + 64) * 32 + soff) * 2);
      unsigned ga  = (unsigned)(((size_t)(m0 + srow) * K + k0 + soff) * 2);
      unsigned ga2 = (unsigned)(((size_t)(m0 + srow + 64) * K + k0 + soff) * 2);
      unsigned gb  = (unsigned)(((size_t)(n0 + srow) * K + k0 + soff) * 2);
      unsigned gb2 = (unsigned)(((size_t)(n0 + srow + 64) * K + k0 + soff) * 2);
      async_copy16(asBase + lofs,  ga,  A);
      async_copy16(asBase + lofs2, ga2, A);
      async_copy16(bsBase + lofs,  gb,  WT);
      async_copy16(bsBase + lofs2, gb2, WT);
    }
    wait_async();
    __syncthreads();

    v16bf a0 = ld_frag(&As[(wr +      ln) * 32 + g * 8]);
    v16bf a1 = ld_frag(&As[(wr + 16 + ln) * 32 + g * 8]);
    #pragma unroll
    for (int ct = 0; ct < 4; ++ct) {
      v16bf b = ld_frag(&Bs[(wc + ct * 16 + ln) * 32 + g * 8]);
      acc[0][ct] = wmma_bf16(a0, b, acc[0][ct]);
      acc[1][ct] = wmma_bf16(a1, b, acc[1][ct]);
    }
  }

  // Epilogue. C layout: row = r + 8*(lane>=16), col = lane&15.
  #pragma unroll
  for (int aidx = 0; aidx < 2; ++aidx) {
    #pragma unroll
    for (int ct = 0; ct < 4; ++ct) {
      int c = n0 + wc + ct * 16 + ln;   // global column
      float bv = bias[c];
      #pragma unroll
      for (int r = 0; r < 8; ++r) {
        int m = m0 + wr + aidx * 16 + r + 8 * g;   // global row
        float val = acc[aidx][ct][r] + bv;
        if (EPI == EPI_F32) {
          outF[(size_t)m * N + c] = val;
        } else if (EPI == EPI_GELU) {
          float ge = 0.5f * val * (1.0f + erff(val * 0.70710678118f));
          outB[(size_t)m * N + c] = (__bf16)ge;
        } else { // EPI_QKV: split into q/k/v with head layouts
          int which = c / 768;
          int d     = c - which * 768;
          int h     = d / 96;
          int dh    = d - h * 96;
          int b     = m >> 10;
          int n     = m & 1023;
          int head  = b * 8 + h;
          if (which == 0)
            qOut[((size_t)head * 1024 + n) * 96 + dh] = (__bf16)(val * 0.1020620726f);
          else if (which == 1)
            kOut[((size_t)head * 1024 + n) * 96 + dh] = (__bf16)val;
          else // v stored transposed: [head][dh][n] so P@V B-frag is contiguous
            vOut[((size_t)head * 96 + dh) * 1024 + n] = (__bf16)val;
        }
      }
    }
  }
}

// ---------------------------------------------------------------------------
// Flash attention with relative-position + type-type bias.
// Grid: (N/128, B*H). Block: 256 threads (8 waves), wave = 16 query rows.
// q: [BH][N][96] bf16 (pre-scaled by 1/sqrt(Dh)),  k: [BH][N][96] bf16
// vT: [BH][96][N] bf16,  y: [B][N][768] bf16 (heads re-interleaved)
// ---------------------------------------------------------------------------
__global__ __launch_bounds__(256) void attn_flash_wmma(
    const __bf16* __restrict__ qb, const __bf16* __restrict__ kb,
    const __bf16* __restrict__ vtb, const float* __restrict__ centers,
    const int* __restrict__ types, const int* __restrict__ node_mask,
    const float* __restrict__ rel_emb, const float* __restrict__ tt,
    __bf16* __restrict__ yb) {
  const int bh = blockIdx.y;
  const int bb = bh >> 3;
  const int h  = bh & 7;
  const int q0 = blockIdx.x * 128;

  const int tid  = threadIdx.x;
  const int wave = tid >> 5;
  const int lane = tid & 31;
  const int g    = lane >> 4;
  const int ln   = lane & 15;

  __shared__ __bf16 Kt[32 * 96];      // [key][dh]
  __shared__ __bf16 Vt[96 * 32];      // [dh][key] (already transposed in memory)
  __shared__ __bf16 Pt[8][16 * 32];   // per-wave P tile [row][key]
  __shared__ float cq[128];
  __shared__ int   tq[128];
  __shared__ float ck[32];
  __shared__ int   tk[32];
  __shared__ int   mk[32];
  __shared__ float relh[41];
  __shared__ float tth[4];

  const unsigned ktBase = (unsigned)(uintptr_t)&Kt[0];
  const unsigned vtBase = (unsigned)(uintptr_t)&Vt[0];

  if (tid < 128) {
    cq[tid] = centers[bb * 1024 + q0 + tid];
    tq[tid] = types[bb * 1024 + q0 + tid];
  }
  if (tid < 41) relh[tid] = rel_emb[tid * 8 + h];
  if (tid < 4)  tth[tid]  = tt[h * 4 + tid];

  // Q fragments held in registers for the whole pass (Dh=96 -> 3 K-steps)
  const int qrow = q0 + wave * 16 + ln;
  const __bf16* qbase = qb + ((size_t)bh * 1024 + qrow) * 96;
  v16bf qf[3];
  #pragma unroll
  for (int f = 0; f < 3; ++f) qf[f] = ld_frag(qbase + f * 32 + g * 8);

  v8f o[6] = {};
  float rmax[8], rsum[8];
  #pragma unroll
  for (int r = 0; r < 8; ++r) { rmax[r] = -3.0e38f; rsum[r] = 0.0f; }

  for (int k0 = 0; k0 < 1024; k0 += 32) {
    __syncthreads();
    // Async-stage K tile 32x96 and V^T tile 96x32 (384 x 16B chunks each)
    for (int i = tid; i < 384; i += 256) {
      int row = i / 12, c = (i % 12) * 8;
      unsigned lofs = (unsigned)((row * 96 + c) * 2);
      unsigned gofs = (unsigned)((((size_t)bh * 1024 + k0 + row) * 96 + c) * 2);
      async_copy16(ktBase + lofs, gofs, kb);
    }
    for (int i = tid; i < 384; i += 256) {
      int row = i / 4, c = (i % 4) * 8;
      unsigned lofs = (unsigned)((row * 32 + c) * 2);
      unsigned gofs = (unsigned)((((size_t)bh * 96 + row) * 1024 + k0 + c) * 2);
      async_copy16(vtBase + lofs, gofs, vtb);
    }
    if (tid < 32) {
      ck[tid] = centers[bb * 1024 + k0 + tid];
      tk[tid] = types[bb * 1024 + k0 + tid];
      mk[tid] = node_mask[bb * 1024 + k0 + tid];
    }
    wait_async();
    __syncthreads();

    // S = Q * K^T : two 16-key column tiles, 3 WMMAs each over Dh
    v8f s[2] = {};
    #pragma unroll
    for (int ct = 0; ct < 2; ++ct) {
      #pragma unroll
      for (int f = 0; f < 3; ++f) {
        v16bf kf = ld_frag(&Kt[(ct * 16 + ln) * 96 + f * 32 + g * 8]);
        s[ct] = wmma_bf16(qf[f], kf, s[ct]);
      }
    }

    // Bias + mask + online softmax. Row m = r + 8*g lives across a 16-lane
    // half-wave in one accumulator slot -> width-16 butterflies.
    float scl[8];
    #pragma unroll
    for (int r = 0; r < 8; ++r) {
      int rowl = wave * 16 + r + 8 * g;
      float ci = cq[rowl];
      int   ti = tq[rowl];
      float e0, e1;
      {
        int kc = ln;
        float d = fminf(fmaxf(ci - ck[kc], -20.f), 20.f) + 20.f;
        float sv = s[0][r] + relh[(int)rintf(d)] + tth[ti * 2 + tk[kc]];
        e0 = (mk[kc] == 0) ? -1e30f : sv;
      }
      {
        int kc = 16 + ln;
        float d = fminf(fmaxf(ci - ck[kc], -20.f), 20.f) + 20.f;
        float sv = s[1][r] + relh[(int)rintf(d)] + tth[ti * 2 + tk[kc]];
        e1 = (mk[kc] == 0) ? -1e30f : sv;
      }
      float t = fmaxf(e0, e1);
      #pragma unroll
      for (int off = 1; off < 16; off <<= 1) t = fmaxf(t, __shfl_xor(t, off, 16));
      float nm = fmaxf(rmax[r], t);
      float sc = __expf(rmax[r] - nm);
      float p0 = __expf(e0 - nm);
      float p1 = __expf(e1 - nm);
      float ps = p0 + p1;
      #pragma unroll
      for (int off = 1; off < 16; off <<= 1) ps += __shfl_xor(ps, off, 16);
      rsum[r] = rsum[r] * sc + ps;
      rmax[r] = nm;
      scl[r]  = sc;
      // Scatter P into A-layout staging buffer (per-wave, in-order LDS)
      Pt[wave][(r + 8 * g) * 32 + ln]      = (__bf16)p0;
      Pt[wave][(r + 8 * g) * 32 + 16 + ln] = (__bf16)p1;
    }

    // Rescale running output and accumulate P @ V (6 dh column tiles)
    #pragma unroll
    for (int nt = 0; nt < 6; ++nt)
      #pragma unroll
      for (int r = 0; r < 8; ++r) o[nt][r] *= scl[r];

    v16bf pf = ld_frag(&Pt[wave][ln * 32 + g * 8]);
    #pragma unroll
    for (int nt = 0; nt < 6; ++nt) {
      v16bf vf = ld_frag(&Vt[(nt * 16 + ln) * 32 + g * 8]);
      o[nt] = wmma_bf16(pf, vf, o[nt]);
    }
  }

  // Normalize and store y as bf16 with heads re-interleaved: y[b][n][h*96+dh]
  float inv[8];
  #pragma unroll
  for (int r = 0; r < 8; ++r) inv[r] = (rsum[r] > 0.0f) ? (1.0f / rsum[r]) : 0.0f;
  #pragma unroll
  for (int nt = 0; nt < 6; ++nt) {
    #pragma unroll
    for (int r = 0; r < 8; ++r) {
      int nrow = q0 + wave * 16 + r + 8 * g;
      int dh   = nt * 16 + ln;
      yb[((size_t)bb * 1024 + nrow) * 768 + h * 96 + dh] = (__bf16)(o[nt][r] * inv[r]);
    }
  }
}

// ---------------------------------------------------------------------------
// Fused residual-add + LayerNorm. One block per row of D elements.
// Emits f32 (next residual) and optional bf16 (next GEMM A input).
// ---------------------------------------------------------------------------
__global__ __launch_bounds__(256) void add_layernorm(
    const float* __restrict__ a, const float* __restrict__ res,
    const float* __restrict__ gamma, const float* __restrict__ beta,
    float* __restrict__ outF, __bf16* __restrict__ outB, int D) {
  const int row = blockIdx.x;
  const int tid = threadIdx.x;
  __shared__ float red[256];

  float s = 0.f, s2 = 0.f;
  for (int i = tid; i < D; i += 256) {
    float v = a[(size_t)row * D + i] + res[(size_t)row * D + i];
    s += v; s2 += v * v;
  }
  red[tid] = s; __syncthreads();
  for (int o = 128; o > 0; o >>= 1) { if (tid < o) red[tid] += red[tid + o]; __syncthreads(); }
  float mean = red[0] / D; __syncthreads();
  red[tid] = s2; __syncthreads();
  for (int o = 128; o > 0; o >>= 1) { if (tid < o) red[tid] += red[tid + o]; __syncthreads(); }
  float var  = red[0] / D - mean * mean;
  float rstd = rsqrtf(var + 1e-5f);

  for (int i = tid; i < D; i += 256) {
    float v = a[(size_t)row * D + i] + res[(size_t)row * D + i];
    float out = (v - mean) * rstd * gamma[i] + beta[i];
    outF[(size_t)row * D + i] = out;
    if (outB) outB[(size_t)row * D + i] = (__bf16)out;
  }
}

// ---------------------------------------------------------------------------
// Host launch
// ---------------------------------------------------------------------------
extern "C" void kernel_launch(void* const* d_in, const int* in_sizes, int n_in,
                              void* d_out, int out_size, void* d_ws, size_t ws_size,
                              hipStream_t stream) {
  (void)in_sizes; (void)n_in; (void)out_size; (void)ws_size;
  const float* x        = (const float*)d_in[0];
  const int*   nmask    = (const int*)d_in[1];
  const float* centers  = (const float*)d_in[2];
  const int*   types    = (const int*)d_in[3];
  const float* qkv_w    = (const float*)d_in[4];
  const float* qkv_b    = (const float*)d_in[5];
  const float* out_w    = (const float*)d_in[6];
  const float* out_b    = (const float*)d_in[7];
  const float* ff1_w    = (const float*)d_in[8];
  const float* ff1_b    = (const float*)d_in[9];
  const float* ff2_w    = (const float*)d_in[10];
  const float* ff2_b    = (const float*)d_in[11];
  const float* ln1_g    = (const float*)d_in[12];
  const float* ln1_b    = (const float*)d_in[13];
  const float* ln2_g    = (const float*)d_in[14];
  const float* ln2_b    = (const float*)d_in[15];
  const float* rel_emb  = (const float*)d_in[16];
  const float* tt       = (const float*)d_in[17];

  const int B = 8, N = 1024, D = 768, F = 1024;
  const size_t M = (size_t)B * N;           // 8192 rows

  // Workspace carve-out (256B aligned)
  char* ws = (char*)d_ws;
  size_t off = 0;
  auto carve = [&](size_t bytes) { char* p = ws + off; off = (off + bytes + 255) & ~(size_t)255; return p; };
  __bf16* xb    = (__bf16*)carve(M * D * 2);            // x in bf16
  __bf16* qkvT  = (__bf16*)carve((size_t)3 * D * D * 2);// [2304][768]
  __bf16* outT  = (__bf16*)carve((size_t)D * D * 2);    // [768][768]
  __bf16* ff1T  = (__bf16*)carve((size_t)F * D * 2);    // [1024][768]
  __bf16* ff2T  = (__bf16*)carve((size_t)D * F * 2);    // [768][1024]
  __bf16* qbuf  = (__bf16*)carve(M * D * 2);            // [BH][N][96]
  __bf16* kbuf  = (__bf16*)carve(M * D * 2);
  __bf16* vtbuf = (__bf16*)carve(M * D * 2);            // [BH][96][N]
  __bf16* ybuf  = (__bf16*)carve(M * D * 2);            // attn out bf16
  float*  aout  = (float*)carve(M * D * 4);             // out-proj f32
  float*  h1    = (float*)carve(M * D * 4);             // after LN1 f32
  __bf16* h1b   = (__bf16*)carve(M * D * 2);
  __bf16* ffb   = (__bf16*)carve(M * F * 2);            // GELU(ff1) bf16
  float*  ff2o  = (float*)carve(M * D * 4);

  // 1. Conversions
  {
    size_t n = M * D;
    cvt_f32_bf16<<<dim3((unsigned)((n + 255) / 256)), dim3(256), 0, stream>>>(x, xb, n);
  }
  {
    size_t n = (size_t)D * 3 * D;
    transpose_w_bf16<<<dim3((unsigned)((n + 255) / 256)), dim3(256), 0, stream>>>(qkv_w, qkvT, D, 3 * D);
  }
  {
    size_t n = (size_t)D * D;
    transpose_w_bf16<<<dim3((unsigned)((n + 255) / 256)), dim3(256), 0, stream>>>(out_w, outT, D, D);
  }
  {
    size_t n = (size_t)D * F;
    transpose_w_bf16<<<dim3((unsigned)((n + 255) / 256)), dim3(256), 0, stream>>>(ff1_w, ff1T, D, F);
  }
  {
    size_t n = (size_t)F * D;
    transpose_w_bf16<<<dim3((unsigned)((n + 255) / 256)), dim3(256), 0, stream>>>(ff2_w, ff2T, F, D);
  }

  // 2. QKV GEMM: [8192x768] @ [768x2304] -> split q/k/v (v transposed)
  gemm_bf16_wmma<EPI_QKV><<<dim3(64, 18), dim3(256), 0, stream>>>(
      xb, qkvT, qkv_b, (int)M, 3 * D, D, nullptr, nullptr, qbuf, kbuf, vtbuf);

  // 3. Flash attention with biases
  attn_flash_wmma<<<dim3(8, 64), dim3(256), 0, stream>>>(
      qbuf, kbuf, vtbuf, centers, types, nmask, rel_emb, tt, ybuf);

  // 4. Output projection -> f32
  gemm_bf16_wmma<EPI_F32><<<dim3(64, 6), dim3(256), 0, stream>>>(
      ybuf, outT, out_b, (int)M, D, D, aout, nullptr, nullptr, nullptr, nullptr);

  // 5. LN1(x + attn)
  add_layernorm<<<dim3((unsigned)M), dim3(256), 0, stream>>>(
      x, aout, ln1_g, ln1_b, h1, h1b, D);

  // 6. FF1 + GELU -> bf16
  gemm_bf16_wmma<EPI_GELU><<<dim3(64, 8), dim3(256), 0, stream>>>(
      h1b, ff1T, ff1_b, (int)M, F, D, nullptr, ffb, nullptr, nullptr, nullptr);

  // 7. FF2 -> f32
  gemm_bf16_wmma<EPI_F32><<<dim3(64, 6), dim3(256), 0, stream>>>(
      ffb, ff2T, ff2_b, (int)M, D, F, ff2o, nullptr, nullptr, nullptr, nullptr);

  // 8. LN2(h1 + ff) -> final output
  add_layernorm<<<dim3((unsigned)M), dim3(256), 0, stream>>>(
      h1, ff2o, ln2_g, ln2_b, (float*)d_out, nullptr, D);
}
